// GPTEncoder_19576460935993
// MI455X (gfx1250) — matrix operand order repaired and checked by reading, not compiled, not run on hardware
//
#include <hip/hip_runtime.h>
#include <hip/hip_bf16.h>
#include <math.h>

// ---------------------------------------------------------------- constants
#define B_   64
#define G_   512
#define D_   384
#define H_   6
#define L_   12
#define DH_  64
#define TOK_ (B_ * G_)          // 32768
#define D3_  (3 * D_)           // 1152
#define D4_  (4 * D_)           // 1536
#define NEGB (-1e9f)

// ---------------------------------------------------------------- types
typedef __attribute__((ext_vector_type(16))) __bf16 bf16x16;
typedef __attribute__((ext_vector_type(8)))  __bf16 bf16x8;
typedef __attribute__((ext_vector_type(8)))  float  f32x8;

#define WMMA_BF16(a, b, c) \
  __builtin_amdgcn_wmma_f32_16x16x32_bf16(false, (a), false, (b), (short)0, (c), false, false)

// float -> bf16 (round to nearest even), no __bf16 arithmetic needed
static __device__ inline __bf16 f2bf(float f) {
  union { float f; unsigned u; } v; v.f = f;
  unsigned r = v.u + 0x7FFFu + ((v.u >> 16) & 1u);
  unsigned short h = (unsigned short)(r >> 16);
  __bf16 b;
  __builtin_memcpy(&b, &h, 2);
  return b;
}

// xor-butterfly across lanes via ds_swizzle (wave32; mask<16 stays in 16-lane half)
template <int MASK>
static __device__ inline float swz(float x) {
  int i = __builtin_bit_cast(int, x);
  i = __builtin_amdgcn_ds_swizzle(i, (MASK << 10) | 0x1F);
  return __builtin_bit_cast(float, i);
}

static __device__ inline float gelu_exact(float x) {
  return 0.5f * x * (1.0f + erff(x * 0.70710678118654752f));
}

// async global->LDS 16B copy (CDNA5 GLOBAL_LOAD_ASYNC_TO_LDS_B128, tracked by ASYNCcnt)
static __device__ inline void async_copy_b128(const void* gsrc, void* lds_dst) {
  unsigned ldsoff = (unsigned)(unsigned long long)lds_dst;   // low 32 bits = LDS offset
  unsigned long long ga = (unsigned long long)gsrc;
  asm volatile("global_load_async_to_lds_b128 %0, %1, off"
               :: "v"(ldsoff), "v"(ga) : "memory");
}
static __device__ inline void async_wait0() {
  asm volatile("s_wait_asynccnt 0x0" ::: "memory");
}

// ---------------------------------------------------------------- small kernels
__global__ void cvt_f32_bf16_kernel(const float* __restrict__ src,
                                    __bf16* __restrict__ dst, int n) {
  int i = blockIdx.x * 256 + threadIdx.x;
  if (i < n) dst[i] = f2bf(src[i]);
}

// h = concat(start_token(1.0), x[:, :-1, :])  (float4 granularity, D=384 -> 96 float4/row)
__global__ void init_h_kernel(const float* __restrict__ x, float* __restrict__ h) {
  size_t idx = (size_t)blockIdx.x * 256 + threadIdx.x;
  if (idx >= (size_t)TOK_ * 96) return;
  size_t tok = idx / 96;
  int c = (int)(idx % 96);
  int g = (int)(tok & (G_ - 1));
  float4 v;
  if (g == 0) v = make_float4(1.f, 1.f, 1.f, 1.f);
  else        v = ((const float4*)x)[(tok - 1) * 96 + c];
  ((float4*)h)[tok * 96 + c] = v;
}

// pos[b,0,:]=1 ; pos[b,g,:] = gelu(center[b,g-1]@w1+b1)@w2+b2
__global__ __launch_bounds__(128)
void pos_embed_kernel(const float* __restrict__ center,
                      const float* __restrict__ w1, const float* __restrict__ b1,
                      const float* __restrict__ w2, const float* __restrict__ b2,
                      float* __restrict__ pos) {
  __shared__ float hid[128];
  int tok = blockIdx.x;
  int b = tok >> 9, g = tok & (G_ - 1);
  int j = threadIdx.x;
  if (g == 0) {
    for (int d = j; d < D_; d += 128) pos[(size_t)tok * D_ + d] = 1.0f;
    return;
  }
  const float* c = center + ((size_t)b * G_ + (g - 1)) * 3;
  float c0 = c[0], c1 = c[1], c2 = c[2];
  float x = c0 * w1[j] + c1 * w1[128 + j] + c2 * w1[256 + j] + b1[j];
  hid[j] = gelu_exact(x);
  __syncthreads();
  #pragma unroll
  for (int r = 0; r < 3; ++r) {
    int d = j + r * 128;
    float s = b2[d];
    for (int jj = 0; jj < 128; ++jj) s += hid[jj] * w2[jj * D_ + d];
    pos[(size_t)tok * D_ + d] = s;
  }
}

// ------------------------------------------------ random-mask pipeline (3 kernels)
__global__ void mask_stats_kernel(const float* __restrict__ center, int* __restrict__ mst) {
  __shared__ int cnt;
  int b = blockIdx.x;
  if (threadIdx.x == 0) cnt = 0;
  __syncthreads();
  int local = 0;
  for (int g = threadIdx.x; g < G_; g += 256) {
    const float* c = center + ((size_t)b * G_ + g) * 3;
    if (c[0] == -1.f && c[1] == -1.f && c[2] == -1.f) local++;
  }
  atomicAdd(&cnt, local);
  __syncthreads();
  if (threadIdx.x == 0) {
    int n = G_ - cnt;
    int k = (int)((float)n * 0.1f);
    int m = (int)((float)(n - k) * 0.6f);
    mst[b * 3 + 0] = n; mst[b * 3 + 1] = k; mst[b * 3 + 2] = m;
  }
}

__global__ void mask_reduce_kernel(int* __restrict__ mst) {
  if (threadIdx.x == 0) {
    int M = 0;
    for (int b = 0; b < B_; ++b) M = max(M, mst[b * 3 + 2]);
    mst[192] = M;
  }
}

__global__ void mask_final_kernel(const float* __restrict__ center,
                                  const float* __restrict__ noise,
                                  const int* __restrict__ mst,
                                  unsigned* __restrict__ colmask,
                                  float* __restrict__ pad_out) {
  __shared__ float u[G_];
  int b = blockIdx.x;
  int n = mst[b * 3], k = mst[b * 3 + 1], m = mst[b * 3 + 2], M = mst[192];
  for (int g = threadIdx.x; g < G_; g += 256) u[g] = noise[(size_t)b * G_ + g];
  __syncthreads();
  for (int i = threadIdx.x; i < G_; i += 256) {
    bool rnd = false;
    if (i >= k && i < n) {
      float ui = u[i]; int rank = 0;
      for (int jj = k; jj < n; ++jj) {
        float uj = u[jj];
        rank += (uj < ui) || (uj == ui && jj < i);
      }
      rnd = rank < m;
    }
    bool tail = (i >= n) && (i < n + M - m);
    rnd = rnd || tail;
    const float* c = center + ((size_t)b * G_ + i) * 3;
    bool pad = (c[0] == -1.f && c[1] == -1.f && c[2] == -1.f);
    colmask[(size_t)b * G_ + i] = (rnd ? 1u : 0u) | (pad ? 2u : 0u);
    pad_out[(size_t)b * G_ + i] = pad ? 1.f : 0.f;
  }
}

// attn_mask output: [B*H, G, G] floats; mask = (k>q) | (rand_k & k!=q)  (pad excluded)
__global__ __launch_bounds__(128)
void mask_out_kernel(const unsigned* __restrict__ colmask, float* __restrict__ mout) {
  int row = blockIdx.x;                 // B*H*G rows
  int q = row & (G_ - 1);
  int b = row / (H_ * G_);
  const unsigned* cm = colmask + (size_t)b * G_;
  int k0 = threadIdx.x * 4;
  float4 v;
  { int kk = k0 + 0; v.x = ((kk > q) || ((cm[kk] & 1u) && kk != q)) ? 1.f : 0.f; }
  { int kk = k0 + 1; v.y = ((kk > q) || ((cm[kk] & 1u) && kk != q)) ? 1.f : 0.f; }
  { int kk = k0 + 2; v.z = ((kk > q) || ((cm[kk] & 1u) && kk != q)) ? 1.f : 0.f; }
  { int kk = k0 + 3; v.w = ((kk > q) || ((cm[kk] & 1u) && kk != q)) ? 1.f : 0.f; }
  *(float4*)(mout + (size_t)row * G_ + k0) = v;
}

// ------------------------------------------------ LayerNorm (wave per token)
// MODE 0: h += pos (in place), write LN -> bf16
// MODE 1: write LN(h) -> bf16
// MODE 2: write LN(h) -> f32 (final output)
template <int MODE>
__global__ __launch_bounds__(256)
void ln_kernel(float* __restrict__ h, const float* __restrict__ pos,
               const float* __restrict__ gamma, const float* __restrict__ beta,
               __bf16* __restrict__ hb, float* __restrict__ hf) {
  int wave = threadIdx.x >> 5, lane = threadIdx.x & 31;
  size_t tok = (size_t)blockIdx.x * 8 + wave;
  float* hp = h + tok * D_;
  float v[12];
  float s = 0.f;
  #pragma unroll
  for (int j = 0; j < 12; ++j) {
    int idx = lane + j * 32;
    float x = hp[idx];
    if (MODE == 0) { x += pos[tok * D_ + idx]; hp[idx] = x; }
    v[j] = x; s += x;
  }
  s += swz<1>(s); s += swz<2>(s); s += swz<4>(s); s += swz<8>(s); s += swz<16>(s);
  float mu = s * (1.f / (float)D_);
  float q = 0.f;
  #pragma unroll
  for (int j = 0; j < 12; ++j) { float d = v[j] - mu; q += d * d; }
  q += swz<1>(q); q += swz<2>(q); q += swz<4>(q); q += swz<8>(q); q += swz<16>(q);
  float inv = rsqrtf(q * (1.f / (float)D_) + 1e-5f);
  #pragma unroll
  for (int j = 0; j < 12; ++j) {
    int idx = lane + j * 32;
    float y = (v[j] - mu) * inv * gamma[idx] + beta[idx];
    if (MODE == 2) hf[tok * D_ + idx] = y;
    else           hb[tok * D_ + idx] = f2bf(y);
  }
}

// ------------------------------------------------ generic WMMA GEMM
// C[M,N] = act(A[M,K] @ W[K,N] + bias);  RESID: resid += C (f32)   OUTB: outb = bf16(C)
// block = 256 thr (8 waves), tile 128(M) x 64(N), K step 32.
// A tile staged via GLOBAL_LOAD_ASYNC_TO_LDS_B128 (ASYNCcnt path);
// W tile staged transposed through VGPRs (ds_store scatter).
template <bool GELU, bool RESID, bool OUTB>
__global__ __launch_bounds__(256)
void gemm_bf16_kernel(const __bf16* __restrict__ A, const __bf16* __restrict__ W,
                      const float* __restrict__ bias, float* __restrict__ resid,
                      __bf16* __restrict__ outb, int M, int N, int K) {
  __shared__ __align__(16) __bf16 sA[128 * 32];
  __shared__ __align__(16) __bf16 sBt[64 * 32];     // transposed: [n][k]
  const int t = threadIdx.x;
  const int wave = t >> 5, lane = t & 31;
  const int half = lane >> 4, lm = lane & 15;
  const int r0 = blockIdx.y * 128;
  const int c0 = blockIdx.x * 64;

  f32x8 acc[4] = {};
  for (int k0 = 0; k0 < K; k0 += 32) {
    __syncthreads();
    // stage A 128x32 via async global->LDS copies (two 16B copies per thread)
    {
      int idx = t;
      #pragma unroll
      for (int it = 0; it < 2; ++it, idx += 256) {
        int row = idx >> 2, ch = idx & 3;
        async_copy_b128(A + (size_t)(r0 + row) * K + k0 + ch * 8,
                        sA + row * 32 + ch * 8);
      }
    }
    // stage W tile transposed into sBt
    {
      int krow = t >> 3, ch = t & 7;
      bf16x8 v = *(const bf16x8*)(W + (size_t)(k0 + krow) * N + c0 + ch * 8);
      #pragma unroll
      for (int j = 0; j < 8; ++j) sBt[(ch * 8 + j) * 32 + krow] = v[j];
    }
    if (k0 + 32 < K) {
      __builtin_prefetch(W + (size_t)(k0 + 32 + (t >> 3)) * N + c0 + (t & 7) * 8, 0, 3);
    }
    async_wait0();          // async LDS writes visible before the barrier
    __syncthreads();
    // A fragment (wave's 16-row strip)
    bf16x16 af;
    {
      const __bf16* p = sA + (wave * 16 + lm) * 32 + half * 16;
      *(bf16x8*)&af       = *(const bf16x8*)p;
      *((bf16x8*)&af + 1) = *(const bf16x8*)(p + 8);
    }
    #pragma unroll
    for (int ct = 0; ct < 4; ++ct) {
      bf16x16 bfm;
      const __bf16* p = sBt + (ct * 16 + lm) * 32 + half * 16;
      *(bf16x8*)&bfm       = *(const bf16x8*)p;
      *((bf16x8*)&bfm + 1) = *(const bf16x8*)(p + 8);
      acc[ct] = WMMA_BF16(af, bfm, acc[ct]);
    }
  }
  // epilogue
  int rowBase = r0 + wave * 16 + half * 8;
  #pragma unroll
  for (int ct = 0; ct < 4; ++ct) {
    int col = c0 + ct * 16 + lm;
    float bv = bias[col];
    #pragma unroll
    for (int i = 0; i < 8; ++i) {
      float v = acc[ct][i] + bv;
      if (GELU) v = gelu_exact(v);
      size_t o = (size_t)(rowBase + i) * N + col;
      if (RESID) resid[o] += v;
      if (OUTB)  outb[o] = f2bf(v);
    }
  }
}

// ------------------------------------------------ flash attention
// grid (G/64, H, B), block 128 (4 waves), wave owns 16 query rows.
// qkv bf16 rows [tok][3*D] (q at +0, k at +384, v at +768, per head +h*64)
__global__ __launch_bounds__(128)
void attention_kernel(const __bf16* __restrict__ qkv,
                      const unsigned* __restrict__ colmask,
                      __bf16* __restrict__ aout) {
  __shared__ __align__(16) __bf16 sP[4 * 16 * 32];   // per-wave P tile
  __shared__ __align__(16) __bf16 sVt[64 * 32];      // V transposed [feat][key]
  const int qt = blockIdx.x;
  const int hh = blockIdx.y;
  const int b  = blockIdx.z;
  const int t = threadIdx.x, wave = t >> 5, lane = t & 31;
  const int half = lane >> 4, lm = lane & 15;
  const int q0 = qt * 64 + wave * 16;
  const size_t baseTok = (size_t)b * G_;

  // Q fragments (feat 0..31 and 32..63)
  bf16x16 aQ0, aQ1;
  {
    const __bf16* qp = qkv + (baseTok + q0 + lm) * D3_ + hh * DH_;
    *(bf16x8*)&aQ0       = *(const bf16x8*)(qp + half * 16);
    *((bf16x8*)&aQ0 + 1) = *(const bf16x8*)(qp + 8 + half * 16);
    *(bf16x8*)&aQ1       = *(const bf16x8*)(qp + 32 + half * 16);
    *((bf16x8*)&aQ1 + 1) = *(const bf16x8*)(qp + 40 + half * 16);
  }
  f32x8 o0 = {}, o1 = {}, o2 = {}, o3 = {};
  float rm[8], rl[8];
  #pragma unroll
  for (int i = 0; i < 8; ++i) { rm[i] = -1e30f; rl[i] = 0.f; }

  const int nkt = qt * 2 + 2;                      // key tiles of 32, causal horizon
  for (int kt = 0; kt < nkt; ++kt) {
    const int kbase = kt * 32;
    __syncthreads();
    // stage V tile transposed: 32 keys x 64 feats
    {
      int key = t >> 2, ch = t & 3;                // 16 feats per thread
      const __bf16* vp = qkv + (baseTok + kbase + key) * D3_ + 2 * D_ + hh * DH_ + ch * 16;
      bf16x8 v0 = *(const bf16x8*)vp;
      bf16x8 v1 = *(const bf16x8*)(vp + 8);
      #pragma unroll
      for (int j = 0; j < 8; ++j) {
        sVt[(ch * 16 + j) * 32 + key]     = v0[j];
        sVt[(ch * 16 + 8 + j) * 32 + key] = v1[j];
      }
    }
    // S = Q @ K^T for 32 keys (two 16-key fragments, 64-feature contraction)
    f32x8 s0 = {}, s1 = {};
    {
      const __bf16* kp0 = qkv + (baseTok + kbase + lm) * D3_ + D_ + hh * DH_;
      const __bf16* kp1 = kp0 + (size_t)16 * D3_;
      bf16x16 bk;
      *(bf16x8*)&bk     = *(const bf16x8*)(kp0 + half * 16);
      *((bf16x8*)&bk+1) = *(const bf16x8*)(kp0 + half * 16 + 8);
      s0 = WMMA_BF16(aQ0, bk, s0);
      *(bf16x8*)&bk     = *(const bf16x8*)(kp0 + 32 + half * 16);
      *((bf16x8*)&bk+1) = *(const bf16x8*)(kp0 + 32 + half * 16 + 8);
      s0 = WMMA_BF16(aQ1, bk, s0);
      *(bf16x8*)&bk     = *(const bf16x8*)(kp1 + half * 16);
      *((bf16x8*)&bk+1) = *(const bf16x8*)(kp1 + half * 16 + 8);
      s1 = WMMA_BF16(aQ0, bk, s1);
      *(bf16x8*)&bk     = *(const bf16x8*)(kp1 + 32 + half * 16);
      *((bf16x8*)&bk+1) = *(const bf16x8*)(kp1 + 32 + half * 16 + 8);
      s1 = WMMA_BF16(aQ1, bk, s1);
    }
    // scale + bias (causal | (rand & !eye) | pad)
    const int kk0 = kbase + lm, kk1 = kbase + 16 + lm;
    const unsigned cm0 = colmask[(size_t)b * G_ + kk0];
    const unsigned cm1 = colmask[(size_t)b * G_ + kk1];
    float alpha[8];
    #pragma unroll
    for (int i = 0; i < 8; ++i) {
      int q = q0 + half * 8 + i;
      float b0 = ((kk0 > q) || ((cm0 & 1u) && kk0 != q) || (cm0 & 2u)) ? NEGB : 0.f;
      float b1v = ((kk1 > q) || ((cm1 & 1u) && kk1 != q) || (cm1 & 2u)) ? NEGB : 0.f;
      s0[i] = s0[i] * 0.125f + b0;
      s1[i] = s1[i] * 0.125f + b1v;
      // online softmax per row (16-lane half-wave reduction)
      float mx = fmaxf(s0[i], s1[i]);
      mx = fmaxf(mx, swz<1>(mx)); mx = fmaxf(mx, swz<2>(mx));
      mx = fmaxf(mx, swz<4>(mx)); mx = fmaxf(mx, swz<8>(mx));
      float nm = fmaxf(rm[i], mx);
      alpha[i] = __expf(rm[i] - nm);
      rm[i] = nm;
      float p0 = __expf(s0[i] - nm), p1 = __expf(s1[i] - nm);
      s0[i] = p0; s1[i] = p1;
      float ps = p0 + p1;
      ps += swz<1>(ps); ps += swz<2>(ps); ps += swz<4>(ps); ps += swz<8>(ps);
      rl[i] = rl[i] * alpha[i] + ps;
    }
    // write P to wave-private LDS in row-major [16][32]
    {
      __bf16* pw = sP + wave * 512;
      #pragma unroll
      for (int i = 0; i < 8; ++i) {
        int row = half * 8 + i;
        pw[row * 32 + lm]      = f2bf(s0[i]);
        pw[row * 32 + 16 + lm] = f2bf(s1[i]);
      }
    }
    __syncthreads();
    // rescale running O
    #pragma unroll
    for (int i = 0; i < 8; ++i) {
      o0[i] *= alpha[i]; o1[i] *= alpha[i]; o2[i] *= alpha[i]; o3[i] *= alpha[i];
    }
    // P as A fragment
    bf16x16 aP;
    {
      const __bf16* p = sP + wave * 512 + lm * 32 + half * 16;
      *(bf16x8*)&aP       = *(const bf16x8*)p;
      *((bf16x8*)&aP + 1) = *(const bf16x8*)(p + 8);
    }
    // O += P @ V  (4 feature tiles)
    #pragma unroll
    for (int ft = 0; ft < 4; ++ft) {
      bf16x16 bv;
      const __bf16* p = sVt + (ft * 16 + lm) * 32 + half * 16;
      *(bf16x8*)&bv       = *(const bf16x8*)p;
      *((bf16x8*)&bv + 1) = *(const bf16x8*)(p + 8);
      f32x8& oo = (ft == 0 ? o0 : ft == 1 ? o1 : ft == 2 ? o2 : o3);
      oo = WMMA_BF16(aP, bv, oo);
    }
  }
  // normalize + store bf16 [tok][h*64+feat]
  #pragma unroll
  for (int i = 0; i < 8; ++i) {
    float inv = 1.f / (rl[i] + 1e-20f);
    int g = q0 + half * 8 + i;
    __bf16* op = aout + (baseTok + g) * D_ + hh * DH_;
    op[ 0 + lm] = f2bf(o0[i] * inv);
    op[16 + lm] = f2bf(o1[i] * inv);
    op[32 + lm] = f2bf(o2[i] * inv);
    op[48 + lm] = f2bf(o3[i] * inv);
  }
}

// ---------------------------------------------------------------- host launch
extern "C" void kernel_launch(void* const* d_in, const int* in_sizes, int n_in,
                              void* d_out, int out_size, void* d_ws, size_t ws_size,
                              hipStream_t stream) {
  const float* x        = (const float*)d_in[0];
  const float* center   = (const float*)d_in[1];
  const float* noise    = (const float*)d_in[2];
  const float* pos_w1   = (const float*)d_in[3];
  const float* pos_b1   = (const float*)d_in[4];
  const float* pos_w2   = (const float*)d_in[5];
  const float* pos_b2   = (const float*)d_in[6];
  const float* ln1_w    = (const float*)d_in[7];
  const float* ln1_b    = (const float*)d_in[8];
  const float* qkv_w    = (const float*)d_in[9];
  const float* qkv_b    = (const float*)d_in[10];
  const float* proj_w   = (const float*)d_in[11];
  const float* proj_b   = (const float*)d_in[12];
  const float* ln2_w    = (const float*)d_in[13];
  const float* ln2_b    = (const float*)d_in[14];
  const float* fc1_w    = (const float*)d_in[15];
  const float* fc1_b    = (const float*)d_in[16];
  const float* fc2_w    = (const float*)d_in[17];
  const float* fc2_b    = (const float*)d_in[18];
  const float* norm_w   = (const float*)d_in[19];
  const float* norm_b   = (const float*)d_in[20];

  float* out_x    = (float*)d_out;                       // [B,G,D]
  float* out_pad  = out_x + (size_t)TOK_ * D_;           // [B,G]
  float* out_mask = out_pad + (size_t)TOK_;              // [B*H,G,G]

  // workspace layout (bytes)
  char* ws = (char*)d_ws;
  float*    h_buf   = (float*)   (ws);                                    // 50.3 MB
  float*    pos_buf = (float*)   (ws + 50331648);                         // 50.3 MB
  __bf16*   hn_buf  = (__bf16*)  (ws + 100663296);                        // 25.2 MB
  __bf16*   qkv_buf = (__bf16*)  (ws + 125829120);                        // 75.5 MB
  __bf16*   ao_buf  = (__bf16*)  (ws + 201326592);                        // 25.2 MB
  __bf16*   mid_buf = (__bf16*)  (ws + 226492416);                        // 100.7 MB
  __bf16*   wqkv    = (__bf16*)  (ws + 327155712);                        // 10.6 MB
  __bf16*   wproj   = (__bf16*)  (ws + 337772544);                        // 3.5 MB
  __bf16*   wfc1    = (__bf16*)  (ws + 341311488);                        // 14.2 MB
  __bf16*   wfc2    = (__bf16*)  (ws + 355467264);                        // 14.2 MB
  unsigned* colmask = (unsigned*)(ws + 369623040);                        // 128 KB
  int*      mstats  = (int*)     (ws + 369754112);

  // weight conversion (all layers at once)
  const int n_qkv = L_ * D_ * D3_, n_proj = L_ * D_ * D_;
  const int n_fc1 = L_ * D_ * D4_, n_fc2 = L_ * D4_ * D_;
  cvt_f32_bf16_kernel<<<(n_qkv + 255) / 256, 256, 0, stream>>>(qkv_w, wqkv, n_qkv);
  cvt_f32_bf16_kernel<<<(n_proj + 255) / 256, 256, 0, stream>>>(proj_w, wproj, n_proj);
  cvt_f32_bf16_kernel<<<(n_fc1 + 255) / 256, 256, 0, stream>>>(fc1_w, wfc1, n_fc1);
  cvt_f32_bf16_kernel<<<(n_fc2 + 255) / 256, 256, 0, stream>>>(fc2_w, wfc2, n_fc2);

  // h init (shift + start token), positional embedding
  init_h_kernel<<<(TOK_ * 96 + 255) / 256, 256, 0, stream>>>(x, h_buf);
  pos_embed_kernel<<<TOK_, 128, 0, stream>>>(center, pos_w1, pos_b1, pos_w2, pos_b2, pos_buf);

  // random mask pipeline + outputs pad / attn_mask
  mask_stats_kernel<<<B_, 256, 0, stream>>>(center, mstats);
  mask_reduce_kernel<<<1, 32, 0, stream>>>(mstats);
  mask_final_kernel<<<B_, 256, 0, stream>>>(center, noise, mstats, colmask, out_pad);
  mask_out_kernel<<<B_ * H_ * G_, 128, 0, stream>>>(colmask, out_mask);

  const dim3 gemm_grid_qkv (D3_ / 64, TOK_ / 128);
  const dim3 gemm_grid_proj(D_  / 64, TOK_ / 128);
  const dim3 gemm_grid_fc1 (D4_ / 64, TOK_ / 128);
  const dim3 gemm_grid_fc2 (D_  / 64, TOK_ / 128);
  const dim3 attn_grid(G_ / 64, H_, B_);

  for (int l = 0; l < L_; ++l) {
    // h += pos ; hn = LN1(h)
    ln_kernel<0><<<TOK_ / 8, 256, 0, stream>>>(h_buf, pos_buf, ln1_w + l * D_, ln1_b + l * D_,
                                               hn_buf, nullptr);
    // qkv = hn @ Wqkv + b
    gemm_bf16_kernel<false, false, true><<<gemm_grid_qkv, 256, 0, stream>>>(
        hn_buf, wqkv + (size_t)l * D_ * D3_, qkv_b + l * D3_, nullptr, qkv_buf,
        TOK_, D3_, D_);
    // flash attention
    attention_kernel<<<attn_grid, 128, 0, stream>>>(qkv_buf, colmask, ao_buf);
    // h += attn_out @ Wproj + b
    gemm_bf16_kernel<false, true, false><<<gemm_grid_proj, 256, 0, stream>>>(
        ao_buf, wproj + (size_t)l * D_ * D_, proj_b + l * D_, h_buf, nullptr,
        TOK_, D_, D_);
    // hn = LN2(h)
    ln_kernel<1><<<TOK_ / 8, 256, 0, stream>>>(h_buf, nullptr, ln2_w + l * D_, ln2_b + l * D_,
                                               hn_buf, nullptr);
    // mid = gelu(hn @ Wfc1 + b)
    gemm_bf16_kernel<true, false, true><<<gemm_grid_fc1, 256, 0, stream>>>(
        hn_buf, wfc1 + (size_t)l * D_ * D4_, fc1_b + l * D4_, nullptr, mid_buf,
        TOK_, D4_, D_);
    // h += mid @ Wfc2 + b
    gemm_bf16_kernel<false, true, false><<<gemm_grid_fc2, 256, 0, stream>>>(
        mid_buf, wfc2 + (size_t)l * D4_ * D_, fc2_b + l * D_, h_buf, nullptr,
        TOK_, D_, D4_);
  }
  // final LayerNorm -> f32 output
  ln_kernel<2><<<TOK_ / 8, 256, 0, stream>>>(h_buf, nullptr, norm_w, norm_b,
                                             nullptr, out_x);
}
